// StateSpaceSimulator_120259084394
// MI455X (gfx1250) — matrix operand order repaired and checked by reading, not compiled, not run on hardware
//
#include <hip/hip_runtime.h>

typedef __bf16 bf16_t;
typedef __bf16 v16bf __attribute__((ext_vector_type(16)));
typedef __bf16 v8bf  __attribute__((ext_vector_type(8)));
typedef __bf16 v4bf  __attribute__((ext_vector_type(4)));
typedef float  v8f   __attribute__((ext_vector_type(8)));
typedef float  v4f   __attribute__((ext_vector_type(4)));
typedef float  v2f   __attribute__((ext_vector_type(2)));

__device__ __forceinline__ float fast_tanh(float x) {
#if __has_builtin(__builtin_amdgcn_tanhf)
  return __builtin_amdgcn_tanhf(x);
#elif __has_builtin(__builtin_amdgcn_tanh_f32)
  return __builtin_amdgcn_tanh_f32(x);
#else
  float r;
  asm("v_tanh_f32 %0, %1" : "=v"(r) : "v"(x));
  return r;
#endif
}

// Wavefront-scope ordering for same-wave LDS producer->consumer (LDS ops from
// one wave complete in order; this only pins compiler/issue ordering, far
// cheaper than a workgroup s_barrier round-trip).
__device__ __forceinline__ void wave_lds_fence() {
  __builtin_amdgcn_fence(__ATOMIC_ACQ_REL, "wavefront");
}

__device__ __forceinline__ v8f wmma_bf16(v16bf a, v16bf b, v8f c) {
  // D = A(16x32 bf16) * B(32x16 bf16) + C(16x16 f32)
  return __builtin_amdgcn_wmma_f32_16x16x32_bf16(false, a, false, b,
                                                 (short)0, c, false, false);
}

// Build A-operand (16x32 bf16) tile for this lane from a row-major bf16 LDS row.
// Layout: lane<16: e0..15 = K[0..15];
//         lane>=16: e0..15 = K[16..31]   (+koff)
__device__ __forceinline__ v16bf load_a(const bf16_t* row, int koff, int hi) {
  const v8bf a0 = *(const v8bf*)(row + koff + 16 * hi);
  const v8bf a1 = *(const v8bf*)(row + koff + 16 * hi + 8);
  return __builtin_shufflevector(a0, a1, 0, 1, 2, 3, 4, 5, 6, 7,
                                 8, 9, 10, 11, 12, 13, 14, 15);
}

__global__ __launch_bounds__(128) void ssm_rollout(
    const float* __restrict__ x0g, const float* __restrict__ ug,
    const float* __restrict__ thfg, const float* __restrict__ thhg,
    const float* __restrict__ Wf1g, const float* __restrict__ bf1g,
    const float* __restrict__ Wf2g, const float* __restrict__ bf2g,
    const float* __restrict__ Wh1g, const float* __restrict__ bh1g,
    const float* __restrict__ Wh2g, const float* __restrict__ bh2g,
    const float* __restrict__ minxg, const float* __restrict__ maxxg,
    const float* __restrict__ minyg, const float* __restrict__ maxyg,
    float* __restrict__ out) {
  constexpr int B = 512, T = 2048, NX = 32, NU = 8, NY = 16, H = 128, NTH = 4;

  __shared__ __align__(16) bf16_t zf[16][64];   // [x(32) | u(8) | thf(4) | pad]
  __shared__ __align__(16) bf16_t zh[16][64];   // [x(32) | thh(4) | pad]
  __shared__ __align__(16) bf16_t act[16][H];   // hidden activations (bf16)
  __shared__ __align__(16) float  xbuf[16][NX]; // fp32 state carry (clamped)
  __shared__ __align__(16) float  ybuf[16][NY]; // fp32 output carry
  __shared__ __align__(16) float  part[4][16][NX]; // per-wave K-partials
  __shared__ float bf2s[NX], bh2s[NY];

  const int tid  = threadIdx.x;
  const int wv   = tid >> 5;   // wave 0..3: owns hidden cols [32*wv, 32*wv+32)
  const int lane = tid & 31;
  const int nl   = lane & 15;
  const int hi   = lane >> 4;
  const int b0   = blockIdx.x * 16;

  const float xmin = minxg[0], xmax = maxxg[0];
  const float ymin = minyg[0], ymax = maxyg[0];

  // ---------------- init LDS ----------------
  for (int i = tid; i < 16 * 64; i += 128) {
    (&zf[0][0])[i] = (bf16_t)0.0f;
    (&zh[0][0])[i] = (bf16_t)0.0f;
  }
  for (int i = tid; i < 16 * NX; i += 128) {
    const int m = i >> 5, n = i & 31;
    const float v = x0g[(b0 + m) * NX + n];
    xbuf[m][n] = v;
    zf[m][n] = (bf16_t)v;
    zh[m][n] = (bf16_t)v;
  }
  for (int i = tid; i < 16 * NTH; i += 128) {
    const int m = i >> 2, j = i & 3;
    zf[m][40 + j] = (bf16_t)thfg[(b0 + m) * NTH + j];
    zh[m][32 + j] = (bf16_t)thhg[(b0 + m) * NTH + j];
  }
  if (tid < NX) bf2s[tid] = bf2g[tid];
  if (tid < NY) bh2s[tid] = bh2g[tid];

  // ---- weights -> registers once, bf16 B-operand layout (K=16*hi+2j, N=nl) ----
  v16bf Bf1[2][2], Bh1[2][2], Bf2[2], Bh2t;
  float bF1[2], bH1[2];
#pragma unroll
  for (int t2 = 0; t2 < 2; ++t2) {
    const int ncol = 32 * wv + 16 * t2 + nl;
    bF1[t2] = bf1g[ncol];
    bH1[t2] = bh1g[ncol];
#pragma unroll
    for (int c = 0; c < 2; ++c) {
      v16bf bf, bh;
#pragma unroll
      for (int j = 0; j < 8; ++j) {
        const int k = 32 * c + 16 * hi + 2 * j;
        bf[2 * j]     = (k < 44)     ? (bf16_t)Wf1g[k * H + ncol]       : (bf16_t)0.0f;
        bf[2 * j + 1] = (k + 1 < 44) ? (bf16_t)Wf1g[(k + 1) * H + ncol] : (bf16_t)0.0f;
        bh[2 * j]     = (k < 36)     ? (bf16_t)Wh1g[k * H + ncol]       : (bf16_t)0.0f;
        bh[2 * j + 1] = (k + 1 < 36) ? (bf16_t)Wh1g[(k + 1) * H + ncol] : (bf16_t)0.0f;
      }
      Bf1[t2][c] = bf;
      Bh1[t2][c] = bh;
    }
  }
#pragma unroll
  for (int t2 = 0; t2 < 2; ++t2) {
    const int ncol = 16 * t2 + nl;
    v16bf bt;
#pragma unroll
    for (int j = 0; j < 8; ++j) {
      const int k = 32 * wv + 16 * hi + 2 * j;
      bt[2 * j]     = (bf16_t)Wf2g[k * NX + ncol];
      bt[2 * j + 1] = (bf16_t)Wf2g[(k + 1) * NX + ncol];
    }
    Bf2[t2] = bt;
  }
  {
    v16bf bt;
#pragma unroll
    for (int j = 0; j < 8; ++j) {
      const int k = 32 * wv + 16 * hi + 2 * j;
      bt[2 * j]     = (bf16_t)Wh2g[k * NY + nl];
      bt[2 * j + 1] = (bf16_t)Wh2g[(k + 1) * NY + nl];
    }
    Bh2t = bt;
  }

  __syncthreads();

  // ------------- h MLP (shared by y0 and the loop); also stages u[t_next] -----
  auto h_step = [&](bool clampY, int t_next) {
    // h1: this wave's 32 hidden columns (reads zh, writes own act columns)
#pragma unroll
    for (int t2 = 0; t2 < 2; ++t2) {
      v8f c;
#pragma unroll
      for (int r = 0; r < 8; ++r) c[r] = bH1[t2];
      c = wmma_bf16(load_a(&zh[nl][0], 0, hi), Bh1[t2][0], c);
      c = wmma_bf16(load_a(&zh[nl][0], 32, hi), Bh1[t2][1], c);
#pragma unroll
      for (int r = 0; r < 8; ++r)
        act[r + 8 * hi][32 * wv + 16 * t2 + nl] = (bf16_t)fast_tanh(c[r]);
    }
    // h2 reads only this wave's own act columns: same-wave LDS is in-order.
    wave_lds_fence();
    {
      v8f c;
#pragma unroll
      for (int r = 0; r < 8; ++r) c[r] = 0.0f;
      c = wmma_bf16(load_a(&act[nl][0], 32 * wv, hi), Bh2t, c);
#pragma unroll
      for (int r = 0; r < 8; ++r) part[wv][r + 8 * hi][nl] = c[r];
    }
    __syncthreads();
    // combine y (256 elements over 128 threads) + stage u[t_next] into zf
    {
      const int m = tid >> 3, q = tid & 7;
      const int n0 = q * 2;
      v2f y;
#pragma unroll
      for (int e = 0; e < 2; ++e) {
        const int n = n0 + e;
        float yv = part[0][m][n] + part[1][m][n] + part[2][m][n] +
                   part[3][m][n] + bh2s[n];
        if (clampY) yv = fminf(fmaxf(yv, ymin), ymax);
        y[e] = yv;
      }
      *(v2f*)&ybuf[m][n0] = y;
      if (t_next < T) {
        const size_t urow = (size_t)(b0 + m) * T + t_next;
        zf[m][32 + q] = (bf16_t)ug[urow * NU + q];
        if (t_next + 1 < T) __builtin_prefetch(ug + (urow + 1) * NU + q, 0, 1);
      }
    }
    __syncthreads();
  };

  // y carry at t=0 is h(x0), UNclamped (matches reference); also stages u[0]
  h_step(false, 0);

  // ---------------- sequential rollout ----------------
  for (int t = 0; t < T; ++t) {
    // Stream out the carries (x_step, y_step); nontemporal (never re-read).
    // Each thread reads exactly the xbuf/ybuf elements it itself writes later,
    // and zf/zh/act were stable since the last barrier -> no barrier needed.
    {
      const int m = tid >> 3, q = tid & 7;
      const size_t row = (size_t)(b0 + m) * T + t;
      v4f xv = *(const v4f*)&xbuf[m][q * 4];
      __builtin_nontemporal_store(xv, (v4f*)(out + row * NX + q * 4));
      v2f yv = *(const v2f*)&ybuf[m][q * 2];
      __builtin_nontemporal_store(
          yv, (v2f*)(out + (size_t)B * T * NX + row * NY + q * 2));
    }

    // f1: hidden = tanh(z_f @ Wf1 + bf1), this wave's 32 cols
#pragma unroll
    for (int t2 = 0; t2 < 2; ++t2) {
      v8f c;
#pragma unroll
      for (int r = 0; r < 8; ++r) c[r] = bF1[t2];
      c = wmma_bf16(load_a(&zf[nl][0], 0, hi), Bf1[t2][0], c);
      c = wmma_bf16(load_a(&zf[nl][0], 32, hi), Bf1[t2][1], c);
#pragma unroll
      for (int r = 0; r < 8; ++r)
        act[r + 8 * hi][32 * wv + 16 * t2 + nl] = (bf16_t)fast_tanh(c[r]);
    }
    // f2 reads only this wave's own act columns: same-wave LDS is in-order.
    wave_lds_fence();
#pragma unroll
    for (int t2 = 0; t2 < 2; ++t2) {
      v8f c;
#pragma unroll
      for (int r = 0; r < 8; ++r) c[r] = 0.0f;
      c = wmma_bf16(load_a(&act[nl][0], 32 * wv, hi), Bf2[t2], c);
#pragma unroll
      for (int r = 0; r < 8; ++r) part[wv][r + 8 * hi][16 * t2 + nl] = c[r];
    }
    __syncthreads();
    // combine x: x_new = x + dx; h sees pre-clamp x_new; carry is clamped
    {
      const int m = tid >> 3, n0 = (tid & 7) * 4;
      v4f p0 = *(const v4f*)&part[0][m][n0];
      v4f p1 = *(const v4f*)&part[1][m][n0];
      v4f p2 = *(const v4f*)&part[2][m][n0];
      v4f p3 = *(const v4f*)&part[3][m][n0];
      v4f xo = *(const v4f*)&xbuf[m][n0];
      v4bf zhw, zfw;
      v4f xc;
#pragma unroll
      for (int e = 0; e < 4; ++e) {
        const float dx = p0[e] + p1[e] + p2[e] + p3[e] + bf2s[n0 + e];
        const float xn = xo[e] + dx;
        zhw[e] = (bf16_t)xn;  // pre-clamp for h
        const float xcl = fminf(fmaxf(xn, xmin), xmax);
        xc[e] = xcl;
        zfw[e] = (bf16_t)xcl;
      }
      *(v4f*)&xbuf[m][n0] = xc;
      *(v4bf*)&zh[m][n0] = zhw;
      *(v4bf*)&zf[m][n0] = zfw;
    }
    __syncthreads();

    h_step(true, t + 1);
  }
}

extern "C" void kernel_launch(void* const* d_in, const int* in_sizes, int n_in,
                              void* d_out, int out_size, void* d_ws,
                              size_t ws_size, hipStream_t stream) {
  (void)in_sizes; (void)n_in; (void)out_size; (void)d_ws; (void)ws_size;
  ssm_rollout<<<dim3(512 / 16), dim3(128), 0, stream>>>(
      (const float*)d_in[0], (const float*)d_in[1], (const float*)d_in[2],
      (const float*)d_in[3], (const float*)d_in[4], (const float*)d_in[5],
      (const float*)d_in[6], (const float*)d_in[7], (const float*)d_in[8],
      (const float*)d_in[9], (const float*)d_in[10], (const float*)d_in[11],
      (const float*)d_in[12], (const float*)d_in[13], (const float*)d_in[14],
      (const float*)d_in[15], (float*)d_out);
}
